// SelfAttention_74689481277587
// MI455X (gfx1250) — matrix operand order repaired and checked by reading, not compiled, not run on hardware
//
#include <hip/hip_runtime.h>

// ---------- CDNA5 WMMA types ----------
typedef __attribute__((ext_vector_type(16))) __bf16        v16bf;
typedef __attribute__((ext_vector_type(8)))  float         v8f;
typedef __attribute__((ext_vector_type(8)))  unsigned int  v8u;
typedef int v4i __attribute__((vector_size(16)));   // pointee type for async-LDS builtin

// ---------- gfx1250 async load-to-LDS (guarded; fallback = sync copy) ----------
#if defined(__AMDGCN__) && defined(__gfx1250__) && __has_builtin(__builtin_amdgcn_global_load_async_to_lds_b128)
  #define HAS_ASYNC_LDS 1
#else
  #define HAS_ASYNC_LDS 0
#endif

static __device__ __forceinline__ void async_wait0() {
#if defined(__AMDGCN__) && defined(__gfx1250__)
  #if __has_builtin(__builtin_amdgcn_s_wait_asynccnt)
    __builtin_amdgcn_s_wait_asynccnt(0);
  #else
    asm volatile("s_wait_asynccnt 0x0" ::: "memory");
  #endif
#endif
}

static __device__ __forceinline__ unsigned short f2bf(float f) {
    union { float f; unsigned u; } x; x.f = f;
    unsigned u = x.u;
    unsigned r = u + 0x7FFFu + ((u >> 16) & 1u);   // round-to-nearest-even
    return (unsigned short)(r >> 16);
}
static __device__ __forceinline__ v8f zero8() {
    v8f z;
#pragma unroll
    for (int i = 0; i < 8; ++i) z[i] = 0.0f;
    return z;
}

// A-matrix 16x32 bf16 fragment (ISA 7.12.2): lane row = lane&15,
// K pairs {hk..hk+7, 16+hk..16+hk+7}, hk = (lane>>4)*8.
static __device__ __forceinline__ v16bf ldsA_frag(const unsigned short* rowp, int hk) {
    v8u u;
    const unsigned* p0 = (const unsigned*)(rowp + hk);
    const unsigned* p1 = (const unsigned*)(rowp + 16 + hk);
#pragma unroll
    for (int j = 0; j < 4; ++j) { u[j] = p0[j]; u[4 + j] = p1[j]; }
    return __builtin_bit_cast(v16bf, u);
}
// B-matrix 32x16 bf16 fragment: lane col = lane&15, K = {kb2/2..kb2/2+7, 16+kb2/2..16+kb2/2+7}, kb2 = (lane>>4)*16.
static __device__ __forceinline__ v16bf ldsB_frag(const unsigned short* rowp, int kb2) {
    v8u u;
    const unsigned* p0 = (const unsigned*)(rowp + (kb2 >> 1));
    const unsigned* p1 = (const unsigned*)(rowp + 16 + (kb2 >> 1));
#pragma unroll
    for (int j = 0; j < 4; ++j) { u[j] = p0[j]; u[4 + j] = p1[j]; }
    return __builtin_bit_cast(v16bf, u);
}

static __device__ __forceinline__ v8f wmma_bf16(v16bf a, v16bf b, v8f c) {
    return __builtin_amdgcn_wmma_f32_16x16x32_bf16(false, a, false, b, (short)0, c, false, false);
}

// ===================== Kernel 1: fused QKV projection -> bf16 =====================
// qb[8192][64]  = (x @ Wq^T) * 0.125   (pre-scaled by 1/sqrt(64))
// kb[8192][64]  =  x @ Wk^T
// vtb[4][64][2048] = (x @ Wv^T) transposed per batch (dim-major)
#define LDW1 40   // 32 + 8 pad (ushorts); 80B row stride (16B aligned, bank-spread)

__global__ __launch_bounds__(256) void qkv_proj_kernel(
    const float* __restrict__ x,
    const float* __restrict__ Wq, const float* __restrict__ Wk, const float* __restrict__ Wv,
    unsigned short* __restrict__ qb, unsigned short* __restrict__ kb,
    unsigned short* __restrict__ vtb)
{
    __shared__ alignas(16) unsigned short xs[64 * LDW1];
    __shared__ alignas(16) unsigned short wsld[192 * LDW1];

    const int t    = threadIdx.x;
    const int lane = t & 31;
    const int w    = t >> 5;
    const int row0 = blockIdx.x * 64;

    const int mtile = w & 3;          // 4 m-tiles of 16 rows
    const int nbase = (w >> 2) * 6;   // each wave: 6 of 12 n-tiles
    const int l15   = lane & 15;
    const int half  = lane >> 4;
    const int hk    = half * 8;
    const int kb2   = half * 16;

    v8f acc[6];
#pragma unroll
    for (int i = 0; i < 6; ++i) acc[i] = zero8();

    for (int kbk = 0; kbk < 1024; kbk += 32) {
        __syncthreads();
#pragma unroll
        for (int i = 0; i < 8; ++i) {
            int idx = t + i * 256;
            int r = idx >> 5, c = idx & 31;
            xs[r * LDW1 + c] = f2bf(x[(size_t)(row0 + r) * 1024 + kbk + c]);
        }
#pragma unroll
        for (int i = 0; i < 24; ++i) {
            int idx = t + i * 256;
            int h = idx >> 5, c = idx & 31;
            const float* Wp = (h < 64) ? Wq : ((h < 128) ? Wk : Wv);
            wsld[h * LDW1 + c] = f2bf(Wp[(size_t)(h & 63) * 1024 + kbk + c]);
        }
        __syncthreads();

        v16bf afrag = ldsA_frag(&xs[(mtile * 16 + l15) * LDW1], hk);
#pragma unroll
        for (int n = 0; n < 6; ++n) {
            int ntile = nbase + n;
            v16bf bfrag = ldsB_frag(&wsld[(ntile * 16 + l15) * LDW1], kb2);
            acc[n] = wmma_bf16(afrag, bfrag, acc[n]);
        }
    }

    const float scale = 0.125f;   // 1/sqrt(64), folded into Q
#pragma unroll
    for (int n = 0; n < 6; ++n) {
        int ntile = nbase + n;
        int buf   = ntile >> 2;                 // 0=q,1=k,2=v
        int hcol  = (ntile & 3) * 16 + l15;
#pragma unroll
        for (int v = 0; v < 8; ++v) {
            int grow = row0 + mtile * 16 + v + 8 * half;   // 0..8191 = b*2048+s
            float val = acc[n][v];
            if (buf == 0) {
                qb[(size_t)grow * 64 + hcol] = f2bf(val * scale);
            } else if (buf == 1) {
                kb[(size_t)grow * 64 + hcol] = f2bf(val);
            } else {
                int bb = grow >> 11, s = grow & 2047;
                vtb[((size_t)bb * 64 + hcol) * 2048 + s] = f2bf(val);
            }
        }
    }
}

// ===================== Kernel 2: flash attention =====================
#define LDK 72    // 64 + 8 pad (ushorts); 144B row stride (16B aligned, bank-spread)

__global__ __launch_bounds__(128) void flash_attn_kernel(
    const unsigned short* __restrict__ qb,
    const unsigned short* __restrict__ kb,
    const unsigned short* __restrict__ vtb,
    const int*   __restrict__ pmask,
    float* __restrict__ out)
{
    __shared__ alignas(16) unsigned short Ks [64 * LDK];    // K tile [key][dim] bf16
    __shared__ alignas(16) unsigned short VTs[64 * LDK];    // V^T tile [dim][key] bf16
    __shared__ alignas(16) unsigned short Ps [4][16 * LDK]; // per-wave P scratch
    __shared__ int pms[64];

    const int t    = threadIdx.x;
    const int lane = t & 31;
    const int w    = t >> 5;
    const int b    = blockIdx.x >> 5;   // 32 q-tiles per batch
    const int qt   = blockIdx.x & 31;
    const int q0   = qt * 64;

    const int l15  = lane & 15;
    const int half = lane >> 4;
    const int hk   = half * 8;
    const int kb2  = half * 16;

    // --- Q fragments straight from bf16 global (already scaled) ---
    const int qrow = q0 + w * 16 + l15;
    v16bf qfrag[2];
    {
        const unsigned short* qp = qb + (size_t)(b * 2048 + qrow) * 64;
#pragma unroll
        for (int c = 0; c < 2; ++c) {
            v8u u;
#pragma unroll
            for (int j = 0; j < 4; ++j) {
                int k0i = c * 32 + hk + 2 * j;
                u[j]     = *(const unsigned*)(qp + k0i);
                u[4 + j] = *(const unsigned*)(qp + k0i + 16);
            }
            qfrag[c] = __builtin_bit_cast(v16bf, u);
        }
    }

    v8f   o[4];
    float m[8], lsum[8];
#pragma unroll
    for (int n = 0; n < 4; ++n) o[n] = zero8();
#pragma unroll
    for (int v = 0; v < 8; ++v) { m[v] = -3.0e38f; lsum[v] = 0.0f; }

    for (int jt = 0; jt <= qt; ++jt) {
        const int k0 = jt * 64;
        __syncthreads();

        // --- tile fill: pure byte copy, 128B contiguous per row for both K and V^T ---
        const unsigned short* kbase = kb  + (size_t)(b * 2048 + k0) * 64;   // [key][64]
        const unsigned short* vbase = vtb + (size_t)b * 64 * 2048 + k0;     // [dim][2048]
#pragma unroll
        for (int i = 0; i < 4; ++i) {
            int idx = t + i * 128;                 // 512 chunks of 16B
            int row = idx >> 3;                    // key (K) / dim (V^T)
            int c8  = (idx & 7) * 8;               // ushort offset within row
            const unsigned short* gk = kbase + (size_t)row * 64   + c8;
            const unsigned short* gv = vbase + (size_t)row * 2048 + c8;
            unsigned short* lk = &Ks [row * LDK + c8];
            unsigned short* lv = &VTs[row * LDK + c8];
#if HAS_ASYNC_LDS
            __builtin_amdgcn_global_load_async_to_lds_b128((v4i*)gk, (v4i*)lk, 0, 0);
            __builtin_amdgcn_global_load_async_to_lds_b128((v4i*)gv, (v4i*)lv, 0, 0);
#else
            *(uint4*)lk = *(const uint4*)gk;
            *(uint4*)lv = *(const uint4*)gv;
#endif
        }
        if (t < 64) pms[t] = pmask[b * 2048 + k0 + t];
        if (jt < qt && t < 64) {   // prefetch next tiles (global_prefetch_b8)
            __builtin_prefetch((const char*)(kbase + 64 * 64) + t * 128, 0, 1);
            __builtin_prefetch((const char*)(vbase + (size_t)t * 2048 + 64), 0, 1);
        }
#if HAS_ASYNC_LDS
        async_wait0();             // ASYNCcnt==0: this wave's copies landed in LDS
#endif
        __syncthreads();

        // --- scores: S = Q(16x64) . K^T ---
        v8f s[4];
#pragma unroll
        for (int n = 0; n < 4; ++n) {
            const unsigned short* krow = &Ks[(n * 16 + l15) * LDK];
            v8f a = zero8();
            a = wmma_bf16(qfrag[0], ldsB_frag(krow,      kb2), a);
            a = wmma_bf16(qfrag[1], ldsB_frag(krow + 32, kb2), a);
            s[n] = a;
        }

        // --- mask (causal only on diagonal tile; padding always) ---
        const bool diag = (jt == qt);
#pragma unroll
        for (int n = 0; n < 4; ++n) {
            int kcol = k0 + n * 16 + l15;
            int pm   = pms[n * 16 + l15];
#pragma unroll
            for (int v = 0; v < 8; ++v) {
                int qr = q0 + w * 16 + v + 8 * half;
                if (pm == 0 || (diag && kcol > qr)) s[n][v] = -1.0e9f;
            }
        }

        // --- online softmax (row reductions across 16-lane halves) ---
#pragma unroll
        for (int v = 0; v < 8; ++v) {
            float mx = fmaxf(fmaxf(s[0][v], s[1][v]), fmaxf(s[2][v], s[3][v]));
#pragma unroll
            for (int off = 1; off < 16; off <<= 1) mx = fmaxf(mx, __shfl_xor(mx, off, 32));
            float mn   = fmaxf(m[v], mx);
            float corr = __expf(m[v] - mn);
            m[v] = mn;
            float rs = 0.0f;
#pragma unroll
            for (int n = 0; n < 4; ++n) {
                float p = __expf(s[n][v] - mn);
                s[n][v] = p;
                rs += p;
            }
#pragma unroll
            for (int off = 1; off < 16; off <<= 1) rs += __shfl_xor(rs, off, 32);
            lsum[v] = lsum[v] * corr + rs;
#pragma unroll
            for (int n = 0; n < 4; ++n) o[n][v] *= corr;
        }

        // --- P: C-layout -> A-layout via per-wave LDS round trip ---
        unsigned short* Pw = Ps[w];
#pragma unroll
        for (int n = 0; n < 4; ++n)
#pragma unroll
            for (int v = 0; v < 8; ++v)
                Pw[(v + 8 * half) * LDK + n * 16 + l15] = f2bf(s[n][v]);

        // --- O += P(16x64) . V(64x64) ---
#pragma unroll
        for (int c = 0; c < 2; ++c) {
            v16bf pf = ldsA_frag(&Pw[l15 * LDK + c * 32], hk);
#pragma unroll
            for (int n2 = 0; n2 < 4; ++n2) {
                v16bf bv = ldsB_frag(&VTs[(n2 * 16 + l15) * LDK + c * 32], kb2);
                o[n2] = wmma_bf16(pf, bv, o[n2]);
            }
        }
    }

    // --- epilogue: normalize and store f32 ---
#pragma unroll
    for (int v = 0; v < 8; ++v) {
        float inv = (lsum[v] > 0.0f) ? (1.0f / lsum[v]) : 0.0f;
        int grow  = b * 2048 + q0 + w * 16 + v + 8 * half;
#pragma unroll
        for (int n2 = 0; n2 < 4; ++n2)
            out[(size_t)grow * 64 + n2 * 16 + l15] = o[n2][v] * inv;
    }
}

// ===================== launcher =====================
extern "C" void kernel_launch(void* const* d_in, const int* in_sizes, int n_in,
                              void* d_out, int out_size, void* d_ws, size_t ws_size,
                              hipStream_t stream) {
    const float* x  = (const float*)d_in[0];
    const int*   pm = (const int*)  d_in[1];
    const float* Wq = (const float*)d_in[2];
    const float* Wk = (const float*)d_in[3];
    const float* Wv = (const float*)d_in[4];
    float* out = (float*)d_out;

    unsigned short* qb  = (unsigned short*)d_ws;     // 8192*64 bf16 = 1 MB
    unsigned short* kbf = qb  + (size_t)8192 * 64;   // 1 MB
    unsigned short* vtb = kbf + (size_t)8192 * 64;   // 1 MB (dim-major per batch)

    qkv_proj_kernel <<<128, 256, 0, stream>>>(x, Wq, Wk, Wv, qb, kbf, vtb);
    flash_attn_kernel<<<128, 128, 0, stream>>>(qb, kbf, vtb, pm, out);
}
